// QuantLinearDequant_12360915878532
// MI455X (gfx1250) — compile-verified, matrix operand order checked by reading, not run on hardware
//
#include <hip/hip_runtime.h>

// ---------------------------------------------------------------------------
// QuantLinearDequant: Y = X @ (Wq * row_scale/127)^T + bias
//   X [8192,4096] f32, Wq [4096,4096] int8, row_scales/bias [4096] f32
// f16 WMMA GEMM (int8->f16 exact via perm+magic-number, f32 accumulate),
// fused scale/bias epilogue. Compute-bound (AI ~1010 flop/B).
// Software-pipelined, double-buffered LDS, 64x64 wave tiles (16 WMMA/iter).
// ---------------------------------------------------------------------------

typedef __attribute__((ext_vector_type(16))) _Float16 v16h;
typedef __attribute__((ext_vector_type(8)))  _Float16 v8h;
typedef __attribute__((ext_vector_type(8)))  float    v8f;

constexpr int TOKENS = 8192;
constexpr int IN_F   = 4096;
constexpr int OUT_F  = 4096;

constexpr int BM = 128;          // block tile M (tokens)
constexpr int BN = 256;          // block tile N (out features)
constexpr int BK = 32;           // K chunk (f16 WMMA 16x16x32)
constexpr int LDSH = BK + 8;     // padded LDS row stride (80 B): 16 rows of a
                                 // b128 frag load cover all 64 banks exactly once

// CDNA5 16-bit A/B operand layout: lane L -> row (L&15), K-half (L>>4);
// VGPR0-3: K = khalf*8+0..7, VGPR4-7: K = 16+khalf*8+0..7 -> two b128 loads.
__device__ __forceinline__ v16h lds_frag(const _Float16* base, int row, int khalf) {
    const _Float16* p = base + row * LDSH + khalf * 8;     // 16-B aligned
    v8h lo = *(const v8h*)(p);                             // ds_load_b128
    v8h hi = *(const v8h*)(p + 16);                        // ds_load_b128
    return __builtin_shufflevector(lo, hi,
        0, 1, 2, 3, 4, 5, 6, 7, 8, 9, 10, 11, 12, 13, 14, 15);
}

// Exact int8 -> f16 for 8 bytes (2 dwords) in 10 VALU ops:
//   u = w ^ 0x80 in [1,255];  0x6400|u == (f16)(1024+u) exactly;
//   subtract 1152 -> w. All values integers < 2048 -> exact.
__device__ __forceinline__ v8h cvt_i8x8_f16(unsigned wlo, unsigned whi) {
    const unsigned ua = wlo ^ 0x80808080u;
    const unsigned ub = whi ^ 0x80808080u;
    union { unsigned u[4]; v8h h; } r;
    r.u[0] = __builtin_amdgcn_perm(0x64646464u, ua, 0x05010400u); // {64,u1,64,u0}
    r.u[1] = __builtin_amdgcn_perm(0x64646464u, ua, 0x07030602u); // {64,u3,64,u2}
    r.u[2] = __builtin_amdgcn_perm(0x64646464u, ub, 0x05010400u);
    r.u[3] = __builtin_amdgcn_perm(0x64646464u, ub, 0x07030602u);
    v8h bias;
#pragma unroll
    for (int i = 0; i < 8; ++i) bias[i] = (_Float16)(-1152.0f);
    return r.h + bias;                                     // 4x v_pk_add_f16
}

// Per-thread staged data for the next K tile.
struct Stage {
    float4 f0, f1, f2, f3;   // 16 x-floats (half an X row)
    int4   w0, w1;           // 32 w-bytes  (one full W row)
};

__device__ __forceinline__ void stage_load(const float* gx, const signed char* gw,
                                           Stage& s) {
    const float4* px = (const float4*)gx;                  // global_load_b128 x4
    s.f0 = px[0]; s.f1 = px[1]; s.f2 = px[2]; s.f3 = px[3];
    const int4* pw = (const int4*)gw;                      // global_load_b128 x2
    s.w0 = pw[0]; s.w1 = pw[1];
}

__device__ __forceinline__ void stage_store(const Stage& s, _Float16* sx, _Float16* sw) {
    v8h lo, hi;
    lo[0] = (_Float16)s.f0.x; lo[1] = (_Float16)s.f0.y;    // v_cvt_pk_f16_f32 x8
    lo[2] = (_Float16)s.f0.z; lo[3] = (_Float16)s.f0.w;
    lo[4] = (_Float16)s.f1.x; lo[5] = (_Float16)s.f1.y;
    lo[6] = (_Float16)s.f1.z; lo[7] = (_Float16)s.f1.w;
    hi[0] = (_Float16)s.f2.x; hi[1] = (_Float16)s.f2.y;
    hi[2] = (_Float16)s.f2.z; hi[3] = (_Float16)s.f2.w;
    hi[4] = (_Float16)s.f3.x; hi[5] = (_Float16)s.f3.y;
    hi[6] = (_Float16)s.f3.z; hi[7] = (_Float16)s.f3.w;
    *(v8h*)(sx)      = lo;                                 // ds_store_b128 x2
    *(v8h*)(sx + 8)  = hi;

    *(v8h*)(sw)      = cvt_i8x8_f16((unsigned)s.w0.x, (unsigned)s.w0.y);
    *(v8h*)(sw + 8)  = cvt_i8x8_f16((unsigned)s.w0.z, (unsigned)s.w0.w);
    *(v8h*)(sw + 16) = cvt_i8x8_f16((unsigned)s.w1.x, (unsigned)s.w1.y);
    *(v8h*)(sw + 24) = cvt_i8x8_f16((unsigned)s.w1.z, (unsigned)s.w1.w);
}

__device__ __forceinline__ void compute_step(const _Float16* Xb, const _Float16* Wb,
                                             int wm, int wn, int frow, int khalf,
                                             v8f (&acc)[4][4]) {
    v16h af[4];
#pragma unroll
    for (int mi = 0; mi < 4; ++mi)
        af[mi] = lds_frag(Xb, wm * 64 + mi * 16 + frow, khalf);
    v16h bf[4];
#pragma unroll
    for (int ni = 0; ni < 4; ++ni)
        bf[ni] = lds_frag(Wb, wn * 64 + ni * 16 + frow, khalf);
#pragma unroll
    for (int mi = 0; mi < 4; ++mi)
#pragma unroll
        for (int ni = 0; ni < 4; ++ni)
            acc[mi][ni] = __builtin_amdgcn_wmma_f32_16x16x32_f16(
                false, af[mi], false, bf[ni],
                (short)0, acc[mi][ni], false, false);
}

__global__ __launch_bounds__(256)
void qlinear_f16_wmma(const float* __restrict__ x,
                      const signed char* __restrict__ wq,
                      const float* __restrict__ row_scales,
                      const float* __restrict__ bias,
                      float* __restrict__ out) {
    __shared__ _Float16 Xs[2][BM * LDSH];   // 2 x 10 KB
    __shared__ _Float16 Ws[2][BN * LDSH];   // 2 x 20 KB  (total 60 KB)

    const int tid  = threadIdx.x;
    const int lane = tid & 31;
    const int wave = tid >> 5;              // 8 wave32 waves, 2(M) x 4(N)
    const int wm   = wave >> 2;             // wave tile: 64 x 64
    const int wn   = wave & 3;

    const int m0 = blockIdx.y * BM;
    const int n0 = blockIdx.x * BN;

    // X loader: 2 threads per row, 16 floats each. W loader: 1 row per thread.
    const int xrow = tid >> 1;              // 0..127
    const int xcol = (tid & 1) * 16;

    const float*       gx = x  + (size_t)(m0 + xrow) * IN_F + xcol;
    const signed char* gw = wq + (size_t)(n0 + tid) * IN_F;
    const int sxofs = xrow * LDSH + xcol;
    const int swofs = tid * LDSH;

    v8f acc[4][4] = {};
    const int khalf = lane >> 4;
    const int frow  = lane & 15;

    // ---- pipeline prologue: tile 0 -> regs -> LDS[0] ----
    Stage st;
    stage_load(gx, gw, st);
    stage_store(st, &Xs[0][sxofs], &Ws[0][swofs]);
    __syncthreads();

    int cur = 0;
    for (int k0 = 0; k0 < IN_F; k0 += BK) {
        const bool more = (k0 + BK) < IN_F;
        if (more) {
            stage_load(gx + (k0 + BK), gw + (k0 + BK), st);  // hide behind WMMAs
            if (k0 + 2 * BK < IN_F) {                        // warm L2 2 tiles ahead
                __builtin_prefetch(gx + k0 + 2 * BK, 0, 3);  // global_prefetch_b8
                __builtin_prefetch(gw + k0 + 2 * BK, 0, 3);
            }
        }

        compute_step(Xs[cur], Ws[cur], wm, wn, frow, khalf, acc);  // 16 WMMA

        if (more)
            stage_store(st, &Xs[cur ^ 1][sxofs], &Ws[cur ^ 1][swofs]);
        __syncthreads();                    // one barrier per iteration
        cur ^= 1;
    }

    // ---- epilogue: D layout -> VGPR r: lanes 0-15 M=r, lanes 16-31 M=8+r ----
    const int mbase = m0 + wm * 64 + khalf * 8;
#pragma unroll
    for (int ni = 0; ni < 4; ++ni) {
        const int n = n0 + wn * 64 + ni * 16 + frow;
        const float s = row_scales[n] * (1.0f / 127.0f);
        const float b = bias[n];
#pragma unroll
        for (int mi = 0; mi < 4; ++mi) {
#pragma unroll
            for (int r = 0; r < 8; ++r) {
                const int m = mbase + mi * 16 + r;
                out[(size_t)m * OUT_F + n] = acc[mi][ni][r] * s + b;
            }
        }
    }
}

extern "C" void kernel_launch(void* const* d_in, const int* in_sizes, int n_in,
                              void* d_out, int out_size, void* d_ws, size_t ws_size,
                              hipStream_t stream) {
    const float*       x      = (const float*)d_in[0];
    const signed char* wq     = (const signed char*)d_in[1];   // int8 payload
    const float*       scales = (const float*)d_in[2];
    const float*       bias   = (const float*)d_in[3];
    float*             out    = (float*)d_out;

    dim3 grid(OUT_F / BN, TOKENS / BM);   // 16 x 64 blocks; N fast-varying
    dim3 block(256);
    qlinear_f16_wmma<<<grid, block, 0, stream>>>(x, wq, scales, bias, out);
}
